// ProposalGenerator_17858474017081
// MI455X (gfx1250) — matrix operand order repaired and checked by reading, not compiled, not run on hardware
//
#include <hip/hip_runtime.h>

typedef __attribute__((ext_vector_type(16))) __bf16 v16bf;
typedef __attribute__((ext_vector_type(8)))  float  v8f;
typedef __attribute__((ext_vector_type(4)))  unsigned int u32x4;
typedef __attribute__((ext_vector_type(4)))  int i32x4;

union Frag { v16bf v; u32x4 q[2]; };

#define CH 256
#define NTOT 261888        // sum over levels of 3*H*W
#define PRE_NMS 2000
#define POST_NMS 1000
#define CAND_CAP 4096
#define DELTA_CAP_C 4.135166556742356f   // log(1000/16)

typedef __attribute__((address_space(1))) i32x4 gi4;   // global int4
typedef __attribute__((address_space(3))) i32x4 li4;   // LDS int4

#if __has_builtin(__builtin_amdgcn_global_load_async_to_lds_b128)
#define ASYNC_LDS 1
#else
#define ASYNC_LDS 0
#endif

// 16-byte global->LDS copy: async DMA on CDNA5, plain copy fallback.
__device__ __forceinline__ void cp16(void* dst, const void* src) {
#if ASYNC_LDS
  __builtin_amdgcn_global_load_async_to_lds_b128(
      (gi4*)(unsigned long long)src,
      (li4*)(unsigned int)(unsigned long long)dst, 0, 0);
#else
  *(u32x4*)dst = *(const u32x4*)src;
#endif
}

__device__ __forceinline__ void async_wait0() {
#if ASYNC_LDS
#if __has_builtin(__builtin_amdgcn_s_wait_asynccnt)
  __builtin_amdgcn_s_wait_asynccnt(0);
#else
  asm volatile("s_wait_asynccnt 0x0" ::: "memory");
#endif
#endif
}

__device__ __forceinline__ unsigned short f2bf(float f) {
  unsigned int u = __float_as_uint(f);
  unsigned int r = u + 0x7FFFu + ((u >> 16) & 1u);   // round-to-nearest-even
  return (unsigned short)(r >> 16);
}

// ---------------- layout conversion / weight packing ----------------
// f32 NCHW -> bf16 NHWC
__global__ __launch_bounds__(256)
void to_nhwc_bf16(const float* __restrict__ in, unsigned short* __restrict__ out, int HW) {
  const int b = blockIdx.z;
  const int pix0 = blockIdx.x * 64;
  const int c = threadIdx.x;
  const float* src = in + (size_t)(b * CH + c) * HW;
  unsigned short* dst = out + ((size_t)b * HW << 8) + c;
  for (int p = 0; p < 64; ++p)
    dst[(size_t)(pix0 + p) << 8] = f2bf(src[pix0 + p]);
}

// conv w OIHW [256][256][3][3] f32 -> bf16 [tap][o][ci] (rows contiguous in ci)
__global__ void pack_w3x3(const float* __restrict__ w, unsigned short* __restrict__ wp) {
  int gid = blockIdx.x * 256 + threadIdx.x;        // < 9*256*256
  int ci  = gid & 255;
  int o   = (gid >> 8) & 255;
  int tap = gid >> 16;
  wp[gid] = f2bf(w[(o * 256 + ci) * 9 + tap]);
}

// heads: obj_w [3][256], delta_w [12][256] -> bf16 [n][ci] (n: 0-2 obj, 3-14 delta, 15 pad)
__global__ void pack_head(const float* __restrict__ ow, const float* __restrict__ dw,
                          unsigned short* __restrict__ whp) {
  int gid = blockIdx.x * 256 + threadIdx.x;        // < 4096
  int ci = gid & 255, n = gid >> 8;
  float v = (n < 3) ? ow[n * 256 + ci] : ((n < 15) ? dw[(n - 3) * 256 + ci] : 0.f);
  whp[gid] = f2bf(v);
}

__global__ void init_zero(float* __restrict__ out, int n, unsigned int* __restrict__ cnt) {
  int i = blockIdx.x * blockDim.x + threadIdx.x;
  if (i < n) out[i] = 0.f;
  if (i < 2) cnt[i] = 0u;
}

// ---------------- 3x3 conv, implicit GEMM, WMMA bf16, 2x2 register blocking ----------------
// WG = 128 thr (4 waves). WG tile: 64 px (M) x 64 och (N); wave tile 32x32 (4 accumulators).
// LDS: A 64 rows x 512B + B 64 rows x 512B, XOR-swizzled 16B chunks (conflict-free b128).
__global__ __launch_bounds__(128)
void conv3x3_wmma(const unsigned short* __restrict__ x,   // NHWC bf16 [b][H][W][256]
                  const unsigned short* __restrict__ wp,  // [tap][o][ci] bf16
                  const float* __restrict__ bias,
                  unsigned short* __restrict__ y,         // NHWC bf16
                  int W, int wshift) {
  __shared__ __align__(16) unsigned char Abuf[64 * 512];
  __shared__ __align__(16) unsigned char Bbuf[64 * 512];

  const int H = W, HW = W << wshift;
  const int b  = blockIdx.z;
  const int p0 = blockIdx.x * 64;
  const int o_base = blockIdx.y * 64;
  const int t  = threadIdx.x;
  const int lane = t & 31, wv = t >> 5;
  const int kh = lane >> 4, lr = lane & 15;
  const int mt2 = wv & 1, nt2 = wv >> 1;

  const int tr = t >> 5;        // staging: row group 0..3
  const int tc = t & 31;        // staging: chunk 0..31

  v8f acc00 = {}, acc01 = {}, acc10 = {}, acc11 = {};
  const u32x4 vzero = {0u, 0u, 0u, 0u};

  for (int tap = 0; tap < 9; ++tap) {
    const int dy = tap / 3 - 1, dx = tap % 3 - 1;

    // ---- stage B: weights [tap][o_base+r][ci], straight async copies
    {
      const unsigned short* wt = wp + (((size_t)tap * 256 + o_base) << 8);
      #pragma unroll
      for (int i = 0; i < 16; ++i) {
        int r = tr + i * 4;
        cp16(Bbuf + r * 512 + (((tc) ^ (r & 15)) << 4), wt + ((size_t)r << 8) + tc * 8);
      }
    }
    // ---- stage A: shifted NHWC rows, zero-fill out-of-image
    #pragma unroll
    for (int i = 0; i < 16; ++i) {
      int r = tr + i * 4;
      int pix = p0 + r;
      int py = (pix >> wshift) + dy, px = (pix & (W - 1)) + dx;
      void* dst = Abuf + r * 512 + (((tc) ^ (r & 15)) << 4);
      if (py >= 0 && py < H && px >= 0 && px < W) {
        const unsigned short* src = x + (((size_t)(b * HW) + (py << wshift) + px) << 8) + tc * 8;
        cp16(dst, src);
      } else {
        *(u32x4*)dst = vzero;
      }
    }
    async_wait0();
    __syncthreads();

    const u32x4* A0 = (const u32x4*)(Abuf + (mt2 * 32 + lr) * 512);
    const u32x4* A1 = (const u32x4*)(Abuf + (mt2 * 32 + 16 + lr) * 512);
    const u32x4* B0 = (const u32x4*)(Bbuf + (nt2 * 32 + lr) * 512);
    const u32x4* B1 = (const u32x4*)(Bbuf + (nt2 * 32 + 16 + lr) * 512);
    #pragma unroll
    for (int kc = 0; kc < 8; ++kc) {
      // A 16x32 bf16: lane holds K {0..7,16..23} (+8 for lanes>=16); B: same K order.
      const int ca0 = (4 * kc + kh) ^ lr, ca1 = (4 * kc + kh + 2) ^ lr;
      const int cb0 = (4 * kc + kh) ^ lr, cb1 = (4 * kc + kh + 2) ^ lr;
      Frag a0, a1, b0, b1;
      a0.q[0] = A0[ca0]; a0.q[1] = A0[ca1];
      a1.q[0] = A1[ca0]; a1.q[1] = A1[ca1];
      b0.q[0] = B0[cb0]; b0.q[1] = B0[cb1];
      b1.q[0] = B1[cb0]; b1.q[1] = B1[cb1];
      acc00 = __builtin_amdgcn_wmma_f32_16x16x32_bf16(false, a0.v, false, b0.v, (short)0, acc00, false, false);
      acc01 = __builtin_amdgcn_wmma_f32_16x16x32_bf16(false, a0.v, false, b1.v, (short)0, acc01, false, false);
      acc10 = __builtin_amdgcn_wmma_f32_16x16x32_bf16(false, a1.v, false, b0.v, (short)0, acc10, false, false);
      acc11 = __builtin_amdgcn_wmma_f32_16x16x32_bf16(false, a1.v, false, b1.v, (short)0, acc11, false, false);
    }
    __syncthreads();
  }

  // ---- epilogue: bias + ReLU, store bf16 NHWC
  const int ch0 = o_base + nt2 * 32;
  const float bv0 = bias[ch0 + lr];
  const float bv1 = bias[ch0 + 16 + lr];
  #pragma unroll
  for (int mi = 0; mi < 2; ++mi) {
    const v8f aa0 = mi ? acc10 : acc00;
    const v8f aa1 = mi ? acc11 : acc01;
    #pragma unroll
    for (int r = 0; r < 8; ++r) {
      int pix = p0 + mt2 * 32 + mi * 16 + kh * 8 + r;
      unsigned short* row = y + (((size_t)(b * HW) + pix) << 8);
      float v0 = aa0[r] + bv0; v0 = v0 > 0.f ? v0 : 0.f;
      float v1 = aa1[r] + bv1; v1 = v1 > 0.f ? v1 : 0.f;
      row[ch0 + lr] = f2bf(v0);
      row[ch0 + 16 + lr] = f2bf(v1);
    }
  }
}

// ---------------- fused 1x1 heads (obj+delta as N=16 GEMM) + anchor decode ----------------
// WG = 128 thr (4 waves), 64 pixels per WG, K=256.
__global__ __launch_bounds__(128)
void head_decode(const unsigned short* __restrict__ t2,   // NHWC bf16
                 const unsigned short* __restrict__ whp,  // [n][ci] bf16
                 const float* __restrict__ obj_b, const float* __restrict__ delta_b,
                 float* __restrict__ boxes, unsigned int* __restrict__ keys,
                 int W, int wshift, int stridep, float asize, int base,
                 const int* __restrict__ image_h, const int* __restrict__ image_w) {
  __shared__ __align__(16) unsigned char Abuf[64 * 512];
  __shared__ __align__(16) unsigned char Bbuf[16 * 512];
  __shared__ float ho[64][16];

  const int HW = W << wshift;
  const int b  = blockIdx.z;
  const int p0 = blockIdx.x * 64;
  const int t  = threadIdx.x;
  const int lane = t & 31, wv = t >> 5;
  const int kh = lane >> 4, lr = lane & 15;
  const int tr = t >> 5, tc = t & 31;

  #pragma unroll
  for (int i = 0; i < 4; ++i) {           // stage B: 16 rows x 32 chunks
    int r = tr + i * 4;
    cp16(Bbuf + r * 512 + ((tc ^ (r & 15)) << 4), whp + ((size_t)r << 8) + tc * 8);
  }
  #pragma unroll
  for (int i = 0; i < 16; ++i) {          // stage A: 64 px x 32 chunks, contiguous NHWC
    int r = tr + i * 4;
    cp16(Abuf + r * 512 + ((tc ^ (r & 15)) << 4),
         t2 + (((size_t)(b * HW) + p0 + r) << 8) + tc * 8);
  }
  async_wait0();
  __syncthreads();

  v8f acc = {};
  const u32x4* A0 = (const u32x4*)(Abuf + (wv * 16 + lr) * 512);
  const u32x4* B0 = (const u32x4*)(Bbuf + lr * 512);
  #pragma unroll
  for (int kc = 0; kc < 8; ++kc) {
    const int ca0 = (4 * kc + kh) ^ lr, ca1 = (4 * kc + kh + 2) ^ lr;
    const int cb0 = (4 * kc + kh) ^ lr, cb1 = (4 * kc + kh + 2) ^ lr;
    Frag a, bb;
    a.q[0]  = A0[ca0]; a.q[1]  = A0[ca1];
    bb.q[0] = B0[cb0]; bb.q[1] = B0[cb1];
    acc = __builtin_amdgcn_wmma_f32_16x16x32_bf16(false, a.v, false, bb.v, (short)0, acc, false, false);
  }
  const int n = lr;
  const float bv = (n < 3) ? obj_b[n] : ((n < 15) ? delta_b[n - 3] : 0.f);
  #pragma unroll
  for (int r = 0; r < 8; ++r) ho[wv * 16 + r + kh * 8][n] = acc[r] + bv;
  __syncthreads();

  const float iw = (float)image_w[0];
  const float ih = (float)image_h[0];
  for (int task = t; task < 192; task += 128) {    // 64 px * 3 anchors
    int ploc = task / 3, a = task - ploc * 3;
    int pix = p0 + ploc;
    int py = pix >> wshift, px = pix & (W - 1);
    float ax = px * (float)stridep + 0.5f * (float)stridep;
    float ay = py * (float)stridep + 0.5f * (float)stridep;
    float s = (a == 0) ? 0.70710678118654752f : ((a == 1) ? 1.f : 1.41421356237309505f);
    float aw = asize * s, ah = asize / s;
    float obj = ho[ploc][a];
    float dxv = ho[ploc][3 + a * 4 + 0];
    float dyv = ho[ploc][3 + a * 4 + 1];
    float dwv = fminf(ho[ploc][3 + a * 4 + 2], DELTA_CAP_C);
    float dhv = fminf(ho[ploc][3 + a * 4 + 3], DELTA_CAP_C);
    float cx = dxv * aw + ax, cy = dyv * ah + ay;
    float w2 = expf(dwv) * aw * 0.5f, h2 = expf(dhv) * ah * 0.5f;
    float x1 = fminf(fmaxf(cx - w2, 0.f), iw);
    float y1 = fminf(fmaxf(cy - h2, 0.f), ih);
    float x2 = fminf(fmaxf(cx + w2, 0.f), iw);
    float y2 = fminf(fmaxf(cy + h2, 0.f), ih);
    long gi = (long)b * NTOT + base + (long)pix * 3 + a;
    float* bp = boxes + gi * 4;
    bp[0] = x1; bp[1] = y1; bp[2] = x2; bp[3] = y2;
    unsigned int u = __float_as_uint(obj);           // sigmoid monotone -> rank on logits
    u ^= (u >> 31) ? 0xFFFFFFFFu : 0x80000000u;      // float -> order-preserving uint
    keys[gi] = u;
  }
}

// ---------------- exact K-th largest key via 4-pass radix histogram ----------------
__global__ __launch_bounds__(256)
void select_thresh(const unsigned int* __restrict__ keys, unsigned int* __restrict__ thr_out,
                   int N, int K) {
  __shared__ unsigned int hist[256];
  __shared__ unsigned int sh_prefix, sh_remaining;
  const int img = blockIdx.x;
  const unsigned int* kk = keys + (long)img * N;
  const int t = threadIdx.x;
  if (t == 0) { sh_prefix = 0u; sh_remaining = (unsigned int)K; }
  __syncthreads();
  for (int pass = 0; pass < 4; ++pass) {
    int shift = 24 - 8 * pass;
    unsigned int mask_hi = (pass == 0) ? 0u : (0xFFFFFFFFu << (shift + 8));
    hist[t] = 0u;
    __syncthreads();
    unsigned int pfx = sh_prefix;
    for (int i = t; i < N; i += 256) {
      unsigned int k = kk[i];
      if ((k & mask_hi) == pfx) atomicAdd(&hist[(k >> shift) & 0xFF], 1u);
    }
    __syncthreads();
    if (t == 0) {
      unsigned int rem = sh_remaining, cum = 0u; int v = 255;
      for (;; --v) { cum += hist[v]; if (cum >= rem || v == 0) break; }
      sh_remaining = rem - (cum - hist[v]);
      sh_prefix = pfx | ((unsigned int)v << shift);
    }
    __syncthreads();
  }
  if (t == 0) thr_out[img] = sh_prefix;
}

__global__ void collect_cands(const unsigned int* __restrict__ keys,
                              const unsigned int* __restrict__ thr,
                              unsigned long long* __restrict__ cand,
                              unsigned int* __restrict__ cnt, int N) {
  int img = blockIdx.y;
  int i = blockIdx.x * blockDim.x + threadIdx.x;
  if (i >= N) return;
  unsigned int k = keys[(long)img * N + i];
  if (k >= thr[img]) {
    unsigned int pos = atomicAdd(&cnt[img], 1u);
    if (pos < (unsigned int)CAND_CAP)
      cand[(long)img * CAND_CAP + pos] = ((unsigned long long)k << 32) | (unsigned int)(~i);
  }
}

// ---------------- bitonic sort 4096 (desc by key, ties: smaller index first) ----------------
__global__ __launch_bounds__(512)
void sort_topk(const unsigned long long* __restrict__ cand, const unsigned int* __restrict__ cnt,
               const float* __restrict__ boxes, float* __restrict__ cand_boxes, int N) {
  __shared__ unsigned long long s[CAND_CAP];
  int img = blockIdx.x;
  int t = threadIdx.x;
  unsigned int c = cnt[img]; if (c > (unsigned int)CAND_CAP) c = CAND_CAP;
  for (int i = t; i < CAND_CAP; i += 512)
    s[i] = (i < (int)c) ? cand[(long)img * CAND_CAP + i] : 0ull;
  __syncthreads();
  for (int ksz = 2; ksz <= CAND_CAP; ksz <<= 1) {
    for (int j = ksz >> 1; j > 0; j >>= 1) {
      for (int i = t; i < CAND_CAP; i += 512) {
        int l = i ^ j;
        if (l > i) {
          unsigned long long a = s[i], b2 = s[l];
          bool up = ((i & ksz) != 0);
          bool sw = up ? (a > b2) : (a < b2);       // overall: descending
          if (sw) { s[i] = b2; s[l] = a; }
        }
      }
      __syncthreads();
    }
  }
  for (int r = t; r < PRE_NMS; r += 512) {
    unsigned long long e = s[r];
    unsigned int idx = ~(unsigned int)(e & 0xFFFFFFFFu);
    float* op = cand_boxes + ((long)img * PRE_NMS + r) * 4;
    if (e != 0ull && idx < (unsigned int)N) {
      const float* bp = boxes + ((long)img * N + idx) * 4;
      op[0] = bp[0]; op[1] = bp[1]; op[2] = bp[2]; op[3] = bp[3];
    } else {
      op[0] = 0.f; op[1] = 0.f; op[2] = 0.f; op[3] = 0.f;
    }
  }
}

// ---------------- greedy NMS (boxes in LDS) + compaction into [1000,4] ----------------
__global__ __launch_bounds__(1024)
void nms_write(const float* __restrict__ cand_boxes, float* __restrict__ out) {
  __shared__ float bx[PRE_NMS][4];
  __shared__ unsigned char keep[PRE_NMS];
  __shared__ unsigned short rnk[PRE_NMS];
  int img = blockIdx.x;
  int t = threadIdx.x;
  for (int i = t; i < PRE_NMS; i += 1024) {
    const float* p = cand_boxes + ((long)img * PRE_NMS + i) * 4;
    bx[i][0] = p[0]; bx[i][1] = p[1]; bx[i][2] = p[2]; bx[i][3] = p[3];
    keep[i] = 1;
  }
  __syncthreads();
  for (int i = 0; i < PRE_NMS; ++i) {
    if (keep[i]) {
      float x1 = bx[i][0], y1 = bx[i][1], x2 = bx[i][2], y2 = bx[i][3];
      float ai = (x2 - x1) * (y2 - y1);
      for (int j = i + 1 + t; j < PRE_NMS; j += 1024) {
        if (!keep[j]) continue;
        float u1 = fmaxf(x1, bx[j][0]), v1 = fmaxf(y1, bx[j][1]);
        float u2 = fminf(x2, bx[j][2]), v2 = fminf(y2, bx[j][3]);
        float iw2 = fmaxf(u2 - u1, 0.f), ih2 = fmaxf(v2 - v1, 0.f);
        float inter = iw2 * ih2;
        float aj = (bx[j][2] - bx[j][0]) * (bx[j][3] - bx[j][1]);
        float iou = inter / fmaxf(ai + aj - inter, 1e-9f);
        if (iou > 0.7f) keep[j] = 0;
      }
    }
    __syncthreads();
  }
  if (t == 0) {
    unsigned int r = 0;
    for (int i = 0; i < PRE_NMS; ++i) { rnk[i] = (unsigned short)r; r += keep[i]; }
  }
  __syncthreads();
  for (int i = t; i < PRE_NMS; i += 1024) {
    if (keep[i] && rnk[i] < POST_NMS) {
      float* op = out + ((long)img * POST_NMS + rnk[i]) * 4;
      op[0] = bx[i][0]; op[1] = bx[i][1]; op[2] = bx[i][2]; op[3] = bx[i][3];
    }
  }
}

// ---------------- host ----------------
extern "C" void kernel_launch(void* const* d_in, const int* in_sizes, int n_in,
                              void* d_out, int out_size, void* d_ws, size_t ws_size,
                              hipStream_t stream) {
  (void)in_sizes; (void)n_in; (void)ws_size;
  const float* conv0_w = (const float*)d_in[5];
  const float* conv0_b = (const float*)d_in[6];
  const float* conv1_w = (const float*)d_in[7];
  const float* conv1_b = (const float*)d_in[8];
  const float* obj_w   = (const float*)d_in[9];
  const float* obj_b   = (const float*)d_in[10];
  const float* delta_w = (const float*)d_in[11];
  const float* delta_b = (const float*)d_in[12];
  const int* image_h   = (const int*)d_in[13];
  const int* image_w   = (const int*)d_in[14];

  char* ws = (char*)d_ws;
  size_t off = 0;
  auto carve = [&](size_t bytes) -> void* {
    void* p = ws + off;
    off += (bytes + 255) & ~(size_t)255;
    return p;
  };
  unsigned short* w0p  = (unsigned short*)carve((size_t)9 * 256 * 256 * 2);
  unsigned short* w1p  = (unsigned short*)carve((size_t)9 * 256 * 256 * 2);
  unsigned short* whp  = (unsigned short*)carve((size_t)16 * 256 * 2);
  unsigned int*   keys = (unsigned int*)carve((size_t)2 * NTOT * 4);
  unsigned int*   thr  = (unsigned int*)carve(2 * 4);
  unsigned int*   cnt  = (unsigned int*)carve(2 * 4);
  unsigned long long* cand = (unsigned long long*)carve((size_t)2 * CAND_CAP * 8);
  float* cand_boxes = (float*)carve((size_t)2 * PRE_NMS * 16);
  float* boxes      = (float*)carve((size_t)2 * NTOT * 16);
  unsigned short* xin = (unsigned short*)carve((size_t)2 * 65536 * 256 * 2);  // NHWC bf16
  unsigned short* t1  = (unsigned short*)carve((size_t)2 * 65536 * 256 * 2);
  unsigned short* t2  = (unsigned short*)carve((size_t)2 * 65536 * 256 * 2);

  init_zero<<<dim3((out_size + 255) / 256), 256, 0, stream>>>((float*)d_out, out_size, cnt);
  pack_w3x3<<<dim3(9 * 256 * 256 / 256), 256, 0, stream>>>(conv0_w, w0p);
  pack_w3x3<<<dim3(9 * 256 * 256 / 256), 256, 0, stream>>>(conv1_w, w1p);
  pack_head<<<dim3(16), 256, 0, stream>>>(obj_w, delta_w, whp);

  const int   Ws[5]      = { 256, 128, 64, 32, 16 };
  const int   wsh[5]     = { 8, 7, 6, 5, 4 };
  const int   strides[5] = { 4, 8, 16, 32, 64 };
  const float sizesl[5]  = { 32.f, 64.f, 128.f, 256.f, 512.f };
  int base = 0;
  for (int l = 0; l < 5; ++l) {
    int W = Ws[l], HW = W * W;
    to_nhwc_bf16<<<dim3(HW / 64, 1, 2), 256, 0, stream>>>((const float*)d_in[l], xin, HW);
    conv3x3_wmma<<<dim3(HW / 64, 4, 2), 128, 0, stream>>>(xin, w0p, conv0_b, t1, W, wsh[l]);
    conv3x3_wmma<<<dim3(HW / 64, 4, 2), 128, 0, stream>>>(t1,  w1p, conv1_b, t2, W, wsh[l]);
    head_decode<<<dim3(HW / 64, 1, 2), 128, 0, stream>>>(t2, whp, obj_b, delta_b, boxes, keys,
                                                         W, wsh[l], strides[l], sizesl[l], base,
                                                         image_h, image_w);
    base += 3 * HW;
  }

  select_thresh<<<dim3(2), 256, 0, stream>>>(keys, thr, NTOT, PRE_NMS);
  collect_cands<<<dim3(NTOT / 256, 2), 256, 0, stream>>>(keys, thr, cand, cnt, NTOT);
  sort_topk<<<dim3(2), 512, 0, stream>>>(cand, cnt, boxes, cand_boxes, NTOT);
  nms_write<<<dim3(2), 1024, 0, stream>>>(cand_boxes, (float*)d_out);
}
